// LeroNet_2138893713985
// MI455X (gfx1250) — matrix operand order repaired and checked by reading, not compiled, not run on hardware
//
#include <hip/hip_runtime.h>
#include <cstdint>
#include <cstddef>

// ---------------------------------------------------------------------------
// LeroNet for MI455X (gfx1250). Heavy math via v_wmma_f32_16x16x32_bf16.
// All operands pre-packed in exact WMMA fragment order so the GEMM hot loop
// does only 2x global_load_b128 per fragment + WMMA. wave32 everywhere.
// ---------------------------------------------------------------------------

typedef __attribute__((ext_vector_type(16))) __bf16        v16bf;
typedef __attribute__((ext_vector_type(8)))  float         v8f;
typedef __attribute__((ext_vector_type(4)))  unsigned int  u32x4;

#define LERO_NEG_SLOPE 0.01f
#define LERO_EPS 1e-5f
#define LERO_B 256
#define LERO_N 1024

union LFrag { v16bf v; u32x4 q[2]; unsigned short u[16]; };

__device__ __forceinline__ unsigned short f2bf(float f) {
    union { float f; unsigned u; } c; c.f = f;
    unsigned u = c.u;
    unsigned r = u + 0x7FFFu + ((u >> 16) & 1u);   // round-to-nearest-even
    return (unsigned short)(r >> 16);
}
__device__ __forceinline__ float bf2f(unsigned short h) {
    union { unsigned u; float f; } c; c.u = ((unsigned)h) << 16;
    return c.f;
}
__device__ __forceinline__ float leaky(float x) {
    return x > 0.f ? x : x * LERO_NEG_SLOPE;
}

// A-fragment K offset for element i within a lane half g (16-bit A 16x32):
//   i in [0,16) -> K = 16g + i (contiguous, matches B packing)
__device__ __forceinline__ int afrag_k(int i, int g) {
    return 16 * g + i;
}

// ---------------------------------------------------------------------------
// Zero the per-batch stats accumulators (graph-safe init every call).
// ---------------------------------------------------------------------------
__global__ void lero_zero_stats(float* stats) {
    stats[threadIdx.x] = 0.f;   // 512 threads, 2 floats per batch
}

// ---------------------------------------------------------------------------
// Pack Wc (co x ci x 3 -> A frags, K = 3ci) and Wm (co x ci -> A frags).
// Layout: [m][s][lane][16] bf16 (32 B per lane -> 2x b128 loads in GEMM).
// ---------------------------------------------------------------------------
template <int CI, int CO>
__global__ __launch_bounds__(256)
void lero_pack_w(const float* __restrict__ Wc, const float* __restrict__ Wm,
                 unsigned short* __restrict__ WcPk,
                 unsigned short* __restrict__ WmPk) {
    constexpr int MT = CO / 16;
    constexpr int SC = (3 * CI) / 32;
    constexpr int SM = CI / 32;
    const int convTot = MT * SC * 32;
    const int mixTot  = MT * SM * 32;
    int tid = blockIdx.x * 256 + threadIdx.x;
    if (tid < convTot) {
        const int lane = tid & 31;
        const int s    = (tid >> 5) % SC;
        const int m    = tid / (32 * SC);
        const int nc = lane & 15, g = lane >> 4;
        const int row = m * 16 + nc;
        LFrag f;
#pragma unroll
        for (int i = 0; i < 16; ++i) {
            int k = s * 32 + afrag_k(i, g);
            int c = k / 3, t = k - 3 * c;
            f.u[i] = f2bf(Wc[((size_t)row * CI + c) * 3 + t]);
        }
        u32x4* o = (u32x4*)(WcPk + (size_t)tid * 16);
        o[0] = f.q[0]; o[1] = f.q[1];
    } else if (tid < convTot + mixTot) {
        const int t2 = tid - convTot;
        const int lane = t2 & 31;
        const int s    = (t2 >> 5) % SM;
        const int m    = t2 / (32 * SM);
        const int nc = lane & 15, g = lane >> 4;
        const int row = m * 16 + nc;
        LFrag f;
#pragma unroll
        for (int i = 0; i < 16; ++i) {
            int k = s * 32 + afrag_k(i, g);
            f.u[i] = f2bf(Wm[(size_t)row * CI + k]);
        }
        u32x4* o = (u32x4*)(WmPk + (size_t)t2 * 16);
        o[0] = f.q[0]; o[1] = f.q[1];
    }
}

// ---------------------------------------------------------------------------
// Pack the combined B matrix in fragment order:
//   rows [0, 3ci)        = gathered triples  x[b][k/3][ idx[b][3n + k%3] ]
//   rows [3ci, 3ci+ci)   = identity (mixer)  x[b][k-3ci][n]
// Layout: P[(b*64 + ntile)*STOT + s][lane][16] bf16.
// B-fragment element i of lane holds K = s*32 + 16*(lane/16) + i (contiguous),
// so each lane's 16 values are one 32 B write, coalesced across lanes.
// ---------------------------------------------------------------------------
template <int CI, bool IN_F32>
__global__ __launch_bounds__(256)
void lero_pack_act(const void* __restrict__ in_,
                   const int* __restrict__ indexes,
                   unsigned short* __restrict__ P) {
    constexpr int KC   = 3 * CI;
    constexpr int STOT = KC / 32 + CI / 32;
    const float*          inF = (const float*)in_;
    const unsigned short* inH = (const unsigned short*)in_;

    size_t tid = (size_t)blockIdx.x * 256 + threadIdx.x;
    if (tid >= (size_t)LERO_B * 64 * STOT * 32) return;
    const int lane = (int)(tid & 31);
    const int s    = (int)((tid >> 5) % STOT);
    const int j    = (int)((tid / (32u * STOT)) % 64);
    const int b    = (int)(tid / ((size_t)32 * STOT * 64));
    const int nc = lane & 15, g = lane >> 4;
    const int n  = j * 16 + nc;
    const int kbase = s * 32 + 16 * g;   // 16 contiguous K values from here

    auto loadX = [&](int c, int nn) -> unsigned short {
        size_t off = ((size_t)b * CI + c) * LERO_N + nn;
        if constexpr (IN_F32) return f2bf(inF[off]);
        else                  return inH[off];
    };

    LFrag f;
    if (kbase < KC) {
        int node[3];
#pragma unroll
        for (int t = 0; t < 3; ++t)
            node[t] = indexes[(size_t)b * (3 * LERO_N) + 3 * n + t];
#pragma unroll
        for (int i = 0; i < 16; ++i) {
            int k = kbase + i;
            int c = k / 3, t = k - 3 * c;
            f.u[i] = loadX(c, node[t]);
        }
    } else {
        const int c0 = kbase - KC;
#pragma unroll
        for (int i = 0; i < 16; ++i)
            f.u[i] = loadX(c0 + i, n);
    }
    u32x4* o = (u32x4*)(P + tid * 16);
    o[0] = f.q[0]; o[1] = f.q[1];
}

// ---------------------------------------------------------------------------
// GEMM: per wave, one 16-node column strip, all CO/16 row tiles resident.
//   Y  = Wc x G (+bc) -> bf16 + fp32 layernorm stats (atomic)
//   XM = Wm x X (+bm) -> bf16
// Hot loop: 2x b128 B-frag load (+ prefetch) then MT x (2x b128 A + WMMA).
// ---------------------------------------------------------------------------
template <int CO, int SC, int SM>
__global__ __launch_bounds__(128)
void lero_gemm(const unsigned short* __restrict__ P,
               const unsigned short* __restrict__ WcPk,
               const unsigned short* __restrict__ WmPk,
               const float* __restrict__ bc, const float* __restrict__ bm,
               unsigned short* __restrict__ outY,
               unsigned short* __restrict__ outXM,
               float* __restrict__ stats) {
    constexpr int MT   = CO / 16;
    constexpr int STOT = SC + SM;
    const int lane = threadIdx.x & 31;
    const int wave = threadIdx.x >> 5;
    const int b    = blockIdx.y;
    const int j    = blockIdx.x * 4 + wave;
    const int nc = lane & 15, g = lane >> 4;
    const int n  = j * 16 + nc;

    const u32x4* Pp = (const u32x4*)(P + (size_t)(b * 64 + j) * STOT * 512);
    const u32x4* Ac = (const u32x4*)WcPk;
    const u32x4* Am = (const u32x4*)WmPk;

    const v8f vzero = {};
    v8f acc[MT];
#pragma unroll
    for (int m = 0; m < MT; ++m) acc[m] = vzero;

    // ---------------- tree-conv GEMM ----------------
    for (int s = 0; s < SC; ++s) {
        LFrag bq;
        bq.q[0] = Pp[s * 64 + lane * 2];
        bq.q[1] = Pp[s * 64 + lane * 2 + 1];
        __builtin_prefetch(&Pp[(s + 1) * 64 + lane * 2], 0, 1);
#pragma unroll
        for (int m = 0; m < MT; ++m) {
            LFrag aq;
            aq.q[0] = Ac[((size_t)(m * SC + s) * 32 + lane) * 2];
            aq.q[1] = Ac[((size_t)(m * SC + s) * 32 + lane) * 2 + 1];
            acc[m] = __builtin_amdgcn_wmma_f32_16x16x32_bf16(
                false, aq.v, false, bq.v, (short)0, acc[m], false, false);
        }
    }

    // Emit Y (bf16) + per-batch sum / sumsq for LayerNorm.
    float s1 = 0.f, s2 = 0.f;
#pragma unroll
    for (int m = 0; m < MT; ++m) {
#pragma unroll
        for (int v = 0; v < 8; ++v) {
            const int row = m * 16 + v + 8 * g;      // C/D: M = v + 8*half
            float y = acc[m][v] + bc[row];
            s1 += y;
            s2 += y * y;
            outY[((size_t)b * CO + row) * LERO_N + n] = f2bf(y);
        }
        acc[m] = vzero;                              // re-arm for mixer
    }
#pragma unroll
    for (int off = 16; off > 0; off >>= 1) {
        s1 += __shfl_xor(s1, off, 32);
        s2 += __shfl_xor(s2, off, 32);
    }
    if (lane == 0) {
        atomicAdd(&stats[2 * b + 0], s1);
        atomicAdd(&stats[2 * b + 1], s2);
    }

    // ---------------- mixer GEMM ----------------
    for (int s = 0; s < SM; ++s) {
        LFrag bq;
        bq.q[0] = Pp[(SC + s) * 64 + lane * 2];
        bq.q[1] = Pp[(SC + s) * 64 + lane * 2 + 1];
#pragma unroll
        for (int m = 0; m < MT; ++m) {
            LFrag aq;
            aq.q[0] = Am[((size_t)(m * SM + s) * 32 + lane) * 2];
            aq.q[1] = Am[((size_t)(m * SM + s) * 32 + lane) * 2 + 1];
            acc[m] = __builtin_amdgcn_wmma_f32_16x16x32_bf16(
                false, aq.v, false, bq.v, (short)0, acc[m], false, false);
        }
    }
#pragma unroll
    for (int m = 0; m < MT; ++m) {
#pragma unroll
        for (int v = 0; v < 8; ++v) {
            const int row = m * 16 + v + 8 * g;
            float xm = acc[m][v] + bm[row];
            outXM[((size_t)b * CO + row) * LERO_N + n] = f2bf(xm);
        }
    }
}

// ---------------------------------------------------------------------------
// Per-batch LayerNorm coefficients: mu and 1/(std_ddof1 + eps).
// ---------------------------------------------------------------------------
__global__ void lero_norm(const float* __restrict__ stats,
                          float* __restrict__ norm, int M) {
    int b = threadIdx.x;
    if (b >= LERO_B) return;
    float s1 = stats[2 * b], s2 = stats[2 * b + 1];
    float mu  = s1 / (float)M;
    float var = (s2 - s1 * s1 / (float)M) / (float)(M - 1);
    var = var > 0.f ? var : 0.f;
    norm[2 * b + 0] = mu;
    norm[2 * b + 1] = 1.f / (sqrtf(var) + LERO_EPS);
}

// ---------------------------------------------------------------------------
// act = leaky_relu(xm + (y - mu) * inv), bf16 out (next block's input).
// ---------------------------------------------------------------------------
__global__ __launch_bounds__(256)
void lero_combine(const unsigned short* __restrict__ Y,
                  const unsigned short* __restrict__ XM,
                  const float* __restrict__ norm,
                  unsigned short* __restrict__ out,
                  int CO, size_t total) {
    size_t i = (size_t)blockIdx.x * blockDim.x + threadIdx.x;
    if (i >= total) return;
    int b = (int)(i / ((size_t)CO * LERO_N));
    float mu  = norm[2 * b + 0];
    float inv = norm[2 * b + 1];
    out[i] = f2bf(leaky(bf2f(XM[i]) + (bf2f(Y[i]) - mu) * inv));
}

// ---------------------------------------------------------------------------
// Masked max-pool over valid nodes + 64->32->1 MLP head. One WG per batch.
// ---------------------------------------------------------------------------
__global__ __launch_bounds__(256)
void lero_head(const unsigned short* __restrict__ X3,
               const int* __restrict__ nodes,
               const float* __restrict__ W1, const float* __restrict__ b1,
               const float* __restrict__ W2, const float* __restrict__ b2,
               float* __restrict__ out) {
    __shared__ float part[256];
    __shared__ float pooled[64];
    __shared__ float h[32];
    const int b   = blockIdx.x;
    const int t   = threadIdx.x;
    const int ch  = t >> 2;
    const int sub = t & 3;
    const int nv  = nodes[b];

    float m = -3.402823466e38f;
    for (int n = sub; n < LERO_N; n += 4) {
        if (n < nv) {
            float v = bf2f(X3[((size_t)b * 64 + ch) * LERO_N + n]);
            m = v > m ? v : m;
        }
    }
    part[t] = m;
    __syncthreads();
    if (t < 64) {
        float mm = part[t * 4];
        mm = fmaxf(mm, part[t * 4 + 1]);
        mm = fmaxf(mm, part[t * 4 + 2]);
        mm = fmaxf(mm, part[t * 4 + 3]);
        pooled[t] = mm;
    }
    __syncthreads();
    if (t < 32) {
        float acc = b1[t];
        for (int c = 0; c < 64; ++c) acc += pooled[c] * W1[c * 32 + t];
        h[t] = leaky(acc);
    }
    __syncthreads();
    if (t == 0) {
        float acc = b2[0];
        for (int j = 0; j < 32; ++j) acc += h[j] * W2[j];
        out[b] = acc;
    }
}

// ---------------------------------------------------------------------------
// Host orchestration (graph-capture safe: only kernel launches on `stream`).
// ---------------------------------------------------------------------------
extern "C" void kernel_launch(void* const* d_in, const int* in_sizes, int n_in,
                              void* d_out, int out_size, void* d_ws, size_t ws_size,
                              hipStream_t stream) {
    const float* trees   = (const float*)d_in[0];
    const int*   indexes = (const int*)d_in[1];
    const int*   nodes   = (const int*)d_in[2];
    const float* Wc1 = (const float*)d_in[3];  const float* bc1 = (const float*)d_in[4];
    const float* Wm1 = (const float*)d_in[5];  const float* bm1 = (const float*)d_in[6];
    const float* Wc2 = (const float*)d_in[7];  const float* bc2 = (const float*)d_in[8];
    const float* Wm2 = (const float*)d_in[9];  const float* bm2 = (const float*)d_in[10];
    const float* Wc3 = (const float*)d_in[11]; const float* bc3 = (const float*)d_in[12];
    const float* Wm3 = (const float*)d_in[13]; const float* bm3 = (const float*)d_in[14];
    const float* W1  = (const float*)d_in[15]; const float* b1  = (const float*)d_in[16];
    const float* W2  = (const float*)d_in[17]; const float* b2  = (const float*)d_in[18];
    float* out = (float*)d_out;

    // Workspace layout (bytes):
    //   [0,2048)         stats
    //   [2048,4096)      norm
    //   [4096, +512K)    WcPk     (max 245 KB)
    //   [.., +512K)      WmPk     (max  82 KB)
    //   [2MB, +512MB)    P        (fragment-ordered combined B, max block2)
    //   then Ybuf (134MB), XMbuf (134MB), actA (134MB), actB (67MB), all bf16
    char* ws = (char*)d_ws;
    float* stats = (float*)ws;
    float* norm  = (float*)(ws + 2048);
    unsigned short* WcPk = (unsigned short*)(ws + 4096);
    unsigned short* WmPk = (unsigned short*)(ws + 4096 + (512 << 10));
    unsigned short* P    = (unsigned short*)(ws + (2ull << 20));
    unsigned short* Ybuf  = (unsigned short*)(ws + (2ull << 20) + (512ull << 20));
    unsigned short* XMbuf = Ybuf  + (size_t)LERO_B * 256 * LERO_N;
    unsigned short* actA  = XMbuf + (size_t)LERO_B * 256 * LERO_N;
    unsigned short* actB  = actA  + (size_t)LERO_B * 256 * LERO_N;

    const dim3 ggrid(LERO_N / 64, LERO_B);  // 4 waves/WG, 16 nodes per wave
    const dim3 gblk(128);

    // ================= block 1: CI=160, CO=256 (fp32 input) =================
    {
        constexpr int CI = 160, CO = 256, SC = 15, SM = 5, STOT = SC + SM;
        int wThreads = (CO / 16) * (SC + SM) * 32;
        lero_pack_w<CI, CO><<<(wThreads + 255) / 256, 256, 0, stream>>>(
            Wc1, Wm1, WcPk, WmPk);
        size_t pThreads = (size_t)LERO_B * 64 * STOT * 32;
        lero_pack_act<CI, true><<<(unsigned)((pThreads + 255) / 256), 256, 0, stream>>>(
            trees, indexes, P);
        lero_zero_stats<<<1, 512, 0, stream>>>(stats);
        lero_gemm<CO, SC, SM><<<ggrid, gblk, 0, stream>>>(
            P, WcPk, WmPk, bc1, bm1, Ybuf, XMbuf, stats);
        lero_norm<<<1, 256, 0, stream>>>(stats, norm, CO * LERO_N);
        size_t tot = (size_t)LERO_B * CO * LERO_N;
        lero_combine<<<(unsigned)((tot + 255) / 256), 256, 0, stream>>>(
            Ybuf, XMbuf, norm, actA, CO, tot);
    }

    // ================= block 2: CI=256, CO=128 (bf16 input) =================
    {
        constexpr int CI = 256, CO = 128, SC = 24, SM = 8, STOT = SC + SM;
        int wThreads = (CO / 16) * (SC + SM) * 32;
        lero_pack_w<CI, CO><<<(wThreads + 255) / 256, 256, 0, stream>>>(
            Wc2, Wm2, WcPk, WmPk);
        size_t pThreads = (size_t)LERO_B * 64 * STOT * 32;
        lero_pack_act<CI, false><<<(unsigned)((pThreads + 255) / 256), 256, 0, stream>>>(
            actA, indexes, P);
        lero_zero_stats<<<1, 512, 0, stream>>>(stats);
        lero_gemm<CO, SC, SM><<<ggrid, gblk, 0, stream>>>(
            P, WcPk, WmPk, bc2, bm2, Ybuf, XMbuf, stats);
        lero_norm<<<1, 256, 0, stream>>>(stats, norm, CO * LERO_N);
        size_t tot = (size_t)LERO_B * CO * LERO_N;
        lero_combine<<<(unsigned)((tot + 255) / 256), 256, 0, stream>>>(
            Ybuf, XMbuf, norm, actB, CO, tot);
    }

    // ================= block 3: CI=128, CO=64 (bf16 input) =================
    {
        constexpr int CI = 128, CO = 64, SC = 12, SM = 4, STOT = SC + SM;
        int wThreads = (CO / 16) * (SC + SM) * 32;
        lero_pack_w<CI, CO><<<(wThreads + 255) / 256, 256, 0, stream>>>(
            Wc3, Wm3, WcPk, WmPk);
        size_t pThreads = (size_t)LERO_B * 64 * STOT * 32;
        lero_pack_act<CI, false><<<(unsigned)((pThreads + 255) / 256), 256, 0, stream>>>(
            actB, indexes, P);
        lero_zero_stats<<<1, 512, 0, stream>>>(stats);
        lero_gemm<CO, SC, SM><<<ggrid, gblk, 0, stream>>>(
            P, WcPk, WmPk, bc3, bm3, Ybuf, XMbuf, stats);
        lero_norm<<<1, 256, 0, stream>>>(stats, norm, CO * LERO_N);
        size_t tot = (size_t)LERO_B * CO * LERO_N;
        lero_combine<<<(unsigned)((tot + 255) / 256), 256, 0, stream>>>(
            Ybuf, XMbuf, norm, actA, CO, tot);
    }

    // ================= head: masked max-pool + MLP =================
    lero_head<<<LERO_B, 256, 0, stream>>>(actA, nodes, W1, b1, W2, b2, out);
}
